// AutoDisLayer_2946347565876
// MI455X (gfx1250) — matrix operand order repaired and checked by reading, not compiled, not run on hardware
//
#include <hip/hip_runtime.h>

#define F_FEAT 128
#define BATCH  4096
#define KB     64
#define EMB    64
#define SLOPE  0.2f

typedef __attribute__((ext_vector_type(16))) __bf16 v16bf;
typedef __attribute__((ext_vector_type(8)))  __bf16 v8bf;
typedef __attribute__((ext_vector_type(8)))  float  v8f;

#define LDK 72   // padded k-stride (bf16 elems) for transposed [n][k] tiles
#define WTS 72   // padded stride for per-wave softmax-weight staging

__device__ __forceinline__ v16bf load_b16x16(const __bf16* p) {
  v8bf lo = *(const v8bf*)(p);
  v8bf hi = *(const v8bf*)(p + 16);
  v16bf r;
#pragma unroll
  for (int j = 0; j < 8; ++j) { r[j] = lo[j]; r[j + 8] = hi[j]; }
  return r;
}

__global__ __launch_bounds__(128, 4)   // cap at 256 VGPRs/wave (no MSB addressing)
void autodis_fused_kernel(const float* __restrict__ dense,
                          const float* __restrict__ w1g,
                          const float* __restrict__ b1g,
                          const float* __restrict__ w2g,
                          const float* __restrict__ b2g,
                          const float* __restrict__ metag,
                          const float* __restrict__ tempg,
                          float* __restrict__ out) {
  __shared__ __bf16 w2t[KB * LDK];    // w2t[n*LDK + k]  = w2[f][k][n]   (bf16)
  __shared__ __bf16 met[EMB * LDK];   // met[d*LDK + k]  = meta[f][k][d] (bf16)
  __shared__ float  w1s[KB], b1s[KB], b2s[KB];
  __shared__ __bf16 wts[4][16 * WTS]; // per-wave softmax weight staging

  const int f    = blockIdx.y;
  const int tid  = threadIdx.x;
  const int wave = tid >> 5;
  const int lane = tid & 31;
  const int lo   = lane & 15;   // N/M position within 16
  const int hi   = lane >> 4;   // half-wave select (K/M split per WMMA layouts)

  // ---- stage per-feature constants into LDS (transposed, bf16) ----
  const float* w2f = w2g   + (size_t)f * KB * KB;
  const float* mef = metag + (size_t)f * KB * EMB;
#pragma unroll
  for (int it = 0; it < 32; ++it) {
    int i = it * 128 + tid;         // i = k*64 + n  (coalesced global reads)
    int k = i >> 6, n = i & 63;
    w2t[n * LDK + k] = (__bf16)w2f[i];
    met[n * LDK + k] = (__bf16)mef[i];
  }
  if (tid < KB) {
    w1s[tid] = w1g[f * KB + tid];
    b1s[tid] = b1g[f * KB + tid];
    b2s[tid] = b2g[f * KB + tid];
  }
  __syncthreads();

  const float invT = 1.0f / tempg[f];

  // Hoist w1/b1 into WMMA 16-bit A-operand order:
  //   j in [0,8):  k = s*32 + hi*8 + j ;  j in [8,16): k = s*32 + 16 + hi*8 + (j-8)
  float w1a[2][16], b1a[2][16];
#pragma unroll
  for (int s = 0; s < 2; ++s)
#pragma unroll
    for (int j = 0; j < 16; ++j) {
      int k = s * 32 + ((j < 8) ? (hi * 8 + j) : (16 + hi * 8 + (j - 8)));
      w1a[s][j] = w1s[k];
      b1a[s][j] = b1s[k];
    }

  float bias[4];
#pragma unroll
  for (int nt = 0; nt < 4; ++nt) bias[nt] = b2s[nt * 16 + lo];

  __bf16* wt = wts[wave];
  const float* xs = dense + (size_t)f * BATCH + blockIdx.x * 256;

#pragma unroll 1
  for (int t2 = 0; t2 < 4; ++t2) {
    const int rb = (t2 * 4 + wave) * 16;          // row-tile base within block
    const float x = xs[rb + lo];                  // this lane's batch row value

    // Opaque zero, redefined every iteration: defeats LICM so the LDS
    // B-operand loads below stay inside the loop (keeps VGPRs < 256).
    int ob = 0;
    asm volatile("" : "+v"(ob));

    // ---- GEMM1: scores = leaky_relu(x*w1+b1) @ w2 ----
    v8f acc[4] = {};
#pragma unroll
    for (int s = 0; s < 2; ++s) {
      v16bf a;
#pragma unroll
      for (int j = 0; j < 16; ++j) {
        float h = fmaf(x, w1a[s][j], b1a[s][j]);
        h = (h > 0.0f) ? h : h * SLOPE;
        a[j] = (__bf16)h;
      }
#pragma unroll
      for (int nt = 0; nt < 4; ++nt) {
        // 16-bit B layout: lane -> n = 16*nt + lo ; element j -> k = s*32 + hi*8 + j (j<8), s*32 + 16 + hi*8 + (j-8) (j>=8)
        v16bf b = load_b16x16(&w2t[ob + (nt * 16 + lo) * LDK + s * 32 + hi * 8]);
        acc[nt] = __builtin_amdgcn_wmma_f32_16x16x32_bf16(
            false, a, false, b, (short)0, acc[nt], false, false);
      }
    }

    // ---- bias + temperature + row softmax over 64 buckets (C/D layout) ----
    // Row M = v + 8*hi lives across lanes of one 16-lane half -> shfl_xor 8/4/2/1.
#pragma unroll
    for (int v = 0; v < 8; ++v) {
      float s0 = (acc[0][v] + bias[0]) * invT;
      float s1 = (acc[1][v] + bias[1]) * invT;
      float s2 = (acc[2][v] + bias[2]) * invT;
      float s3 = (acc[3][v] + bias[3]) * invT;
      float m = fmaxf(fmaxf(s0, s1), fmaxf(s2, s3));
      m = fmaxf(m, __shfl_xor(m, 8, 32));
      m = fmaxf(m, __shfl_xor(m, 4, 32));
      m = fmaxf(m, __shfl_xor(m, 2, 32));
      m = fmaxf(m, __shfl_xor(m, 1, 32));
      s0 = __expf(s0 - m); s1 = __expf(s1 - m);
      s2 = __expf(s2 - m); s3 = __expf(s3 - m);
      float sum = s0 + s1 + s2 + s3;
      sum += __shfl_xor(sum, 8, 32);
      sum += __shfl_xor(sum, 4, 32);
      sum += __shfl_xor(sum, 2, 32);
      sum += __shfl_xor(sum, 1, 32);
      float r = 1.0f / sum;
      acc[0][v] = s0 * r; acc[1][v] = s1 * r;
      acc[2][v] = s2 * r; acc[3][v] = s3 * r;
    }

    // ---- C/D layout -> A layout via per-wave LDS staging (bf16) ----
#pragma unroll
    for (int nt = 0; nt < 4; ++nt)
#pragma unroll
      for (int v = 0; v < 8; ++v)
        wt[(v + 8 * hi) * WTS + nt * 16 + lo] = (__bf16)acc[nt][v];
    // same-wave LDS ops are in order; no barrier needed

    // ---- GEMM2: emb = weights @ meta  (nt-major, store each tile ASAP) ----
    v16bf a2[2];
#pragma unroll
    for (int s = 0; s < 2; ++s) {
      v8bf alo = *(const v8bf*)&wt[lo * WTS + s * 32 + hi * 8];
      v8bf ahi = *(const v8bf*)&wt[lo * WTS + s * 32 + 16 + hi * 8];
#pragma unroll
      for (int j = 0; j < 8; ++j) { a2[s][j] = alo[j]; a2[s][j + 8] = ahi[j]; }
    }

    float* orow = out + (size_t)(blockIdx.x * 256 + rb) * (F_FEAT * EMB) + f * EMB;
#pragma unroll
    for (int nt = 0; nt < 4; ++nt) {
      v8f eacc = {};
#pragma unroll
      for (int s = 0; s < 2; ++s) {
        v16bf b = load_b16x16(&met[ob + (nt * 16 + lo) * LDK + s * 32 + hi * 8]);
        eacc = __builtin_amdgcn_wmma_f32_16x16x32_bf16(
            false, a2[s], false, b, (short)0, eacc, false, false);
      }
#pragma unroll
      for (int v = 0; v < 8; ++v)
        orow[(size_t)(v + 8 * hi) * (F_FEAT * EMB) + nt * 16 + lo] = eacc[v];
    }
  }
}

extern "C" void kernel_launch(void* const* d_in, const int* in_sizes, int n_in,
                              void* d_out, int out_size, void* d_ws, size_t ws_size,
                              hipStream_t stream) {
  (void)in_sizes; (void)n_in; (void)d_ws; (void)ws_size; (void)out_size;
  const float* dense = (const float*)d_in[0];
  const float* w1    = (const float*)d_in[1];
  const float* b1    = (const float*)d_in[2];
  const float* w2    = (const float*)d_in[3];
  const float* b2    = (const float*)d_in[4];
  const float* meta  = (const float*)d_in[5];
  const float* temps = (const float*)d_in[6];
  float* out = (float*)d_out;

  dim3 grid(BATCH / 256, F_FEAT);  // 16 x 128 blocks, 128 threads (4 waves) each
  autodis_fused_kernel<<<grid, 128, 0, stream>>>(dense, w1, b1, w2, b2, meta,
                                                 temps, out);
}
